// TransformerParallelBlock_61349312856280
// MI455X (gfx1250) — compile-verified, matrix operand order checked by reading, not run on hardware
//
#include <hip/hip_runtime.h>
#include <hip/hip_bf16.h>
#include <math.h>

// ---------------- types ----------------
typedef __bf16 bf16_t;
typedef __attribute__((ext_vector_type(16))) __bf16          v16bf;
typedef __attribute__((ext_vector_type(8)))  __bf16          v8bf;
typedef __attribute__((ext_vector_type(8)))  float           v8f;
typedef __attribute__((ext_vector_type(8)))  unsigned short  v8u16;

#define C_DIM   1024
#define H_DIM   16
#define D_DIM   64
#define L_DIM   2048
#define B_DIM   2
#define NROWS   (B_DIM * L_DIM)   // 4096
#define QKV_N   (3 * C_DIM)       // 3072 (dead _mlp_branch of w1 is pruned)
#define CAT_N   (2 * C_DIM)       // 2048

#define KTILE   32                // K-slab per LDS stage (= WMMA K)
#define PADLD   40                // padded LDS row stride (elems): 80B = 20 banks,
                                  // gcd(20,64)=4 -> 16 rows hit 16 distinct 4-bank groups

// ---------------- helpers ----------------
__device__ __forceinline__ bf16_t f2bf(float f) {
  unsigned int u = __builtin_bit_cast(unsigned int, f);
  unsigned int r = u + 0x7FFFu + ((u >> 16) & 1u);   // round-to-nearest-even
  unsigned short h = (unsigned short)(r >> 16);
  return __builtin_bit_cast(bf16_t, h);
}

// Flat shared pointer = {aperture_hi, lds_byte_offset} -> low 32 bits are the LDS address.
__device__ __forceinline__ unsigned lds_addr32(const void* p) {
  return (unsigned)(uintptr_t)p;
}

// Async global->LDS copy of 16 bytes (per lane). Tracked by ASYNCcnt.
__device__ __forceinline__ void async_copy_b128(unsigned lds_off, const void* g) {
  unsigned long long ga = (unsigned long long)(uintptr_t)g;
  asm volatile("global_load_async_to_lds_b128 %0, %1, off"
               :: "v"(lds_off), "v"(ga) : "memory");
}
__device__ __forceinline__ void wait_async0() {
  asm volatile("s_wait_asynccnt 0" ::: "memory");
}

// Load a 16x32 bf16 fragment from GLOBAL (A-layout; B operand of an N x K
// row-major matrix uses the identical per-lane pattern by symmetry).
// lane-lo (0-15): row = lane,    K = k0+{0..7}  and k0+{16..23}
// lane-hi (16-31): row = lane-16, K = k0+{8..15} and k0+{24..31}
__device__ __forceinline__ v16bf load_frag_g(const bf16_t* __restrict__ p,
                                             int row, int k0, int ld, int lane) {
  int r  = row + (lane & 15);
  int ko = k0 + ((lane >> 4) << 3);
  const bf16_t* base = p + (size_t)r * ld + ko;
  v8bf lo = *(const v8bf*)(base);
  v8bf hi = *(const v8bf*)(base + 16);
  v16bf v;
#pragma unroll
  for (int e = 0; e < 8; ++e) { v[e] = lo[e]; v[e + 8] = hi[e]; }
  return v;
}

// Same fragment pattern, but reading a padded LDS tile (row stride PADLD elems).
__device__ __forceinline__ v16bf load_frag_lds(const unsigned short* base0,
                                               int row, int lane) {
  int r  = row + (lane & 15);
  int ko = (lane >> 4) << 3;
  const unsigned short* base = base0 + r * PADLD + ko;
  v8u16 lo = *(const v8u16*)(base);
  v8u16 hi = *(const v8u16*)(base + 16);
  v16bf v;
#pragma unroll
  for (int e = 0; e < 8; ++e) {
    v[e]     = __builtin_bit_cast(bf16_t, (unsigned short)lo[e]);
    v[e + 8] = __builtin_bit_cast(bf16_t, (unsigned short)hi[e]);
  }
  return v;
}

__device__ __forceinline__ v8f wmma_bf16(v16bf a, v16bf b, v8f c) {
  return __builtin_amdgcn_wmma_f32_16x16x32_bf16(false, a, false, b,
                                                 (short)0, c, false, false);
}

// ---------------- kernel 1: LayerNorm + cast ----------------
__global__ __launch_bounds__(256) void ln_kernel(const float* __restrict__ x,
                                                 const float* __restrict__ lw,
                                                 const float* __restrict__ lb,
                                                 bf16_t* __restrict__ h) {
  int row = blockIdx.x;
  int tid = threadIdx.x;              // 256 threads, 4 elems each (C=1024)
  const float* xr = x + (size_t)row * C_DIM;
  float4 v = ((const float4*)xr)[tid];
  float s  = v.x + v.y + v.z + v.w;
  float s2 = v.x * v.x + v.y * v.y + v.z * v.z + v.w * v.w;
#pragma unroll
  for (int m = 1; m < 32; m <<= 1) {
    s  += __shfl_xor(s,  m, 32);
    s2 += __shfl_xor(s2, m, 32);
  }
  __shared__ float red[2][8];
  int wid = tid >> 5, lane = tid & 31;
  if (lane == 0) { red[0][wid] = s; red[1][wid] = s2; }
  __syncthreads();
  s = 0.f; s2 = 0.f;
#pragma unroll
  for (int i = 0; i < 8; ++i) { s += red[0][i]; s2 += red[1][i]; }
  float mu  = s * (1.0f / C_DIM);
  float var = s2 * (1.0f / C_DIM) - mu * mu;
  float inv = rsqrtf(var + 1e-6f);
  bf16_t* hr = h + (size_t)row * C_DIM;
  const float* ve = (const float*)&v;
#pragma unroll
  for (int i = 0; i < 4; ++i) {
    int c = tid * 4 + i;
    hr[c] = f2bf((ve[i] - mu) * inv * lw[c] + lb[c]);
  }
}

// ---------------- kernel 2: f32 -> bf16 cast ----------------
__global__ void cast_bf16_kernel(const float* __restrict__ src,
                                 bf16_t* __restrict__ dst, int n) {
  int i = blockIdx.x * blockDim.x + threadIdx.x;
  if (i < n) dst[i] = f2bf(src[i]);
}

// ---------------- kernel 3: GEMM out[M,N] = A[M,K]*B[N,K]^T (+bias)(+res) ----------------
// 8 waves; workgroup tile 128x128; per-wave 32x64.
// A/B K-slabs staged in LDS via async-to-LDS copies, double-buffered:
//   issue fills(kb+32) -> compute slab(kb) from LDS -> s_wait_asynccnt 0 -> barrier
__global__ __launch_bounds__(256) void gemm_bf16_kernel(
    const bf16_t* __restrict__ A, const bf16_t* __restrict__ Bm,
    const float* __restrict__ bias, const float* __restrict__ residual,
    float* __restrict__ out, int M, int N, int K) {
  __shared__ unsigned short ldsA[2][128 * PADLD];   // 2 x 10240 B
  __shared__ unsigned short ldsB[2][128 * PADLD];   // 2 x 10240 B

  int tid  = threadIdx.x;
  int lane = tid & 31;
  int w    = tid >> 5;
  int wm   = w & 3;                 // 4 M-subtiles of 32
  int wn   = w >> 2;                // 2 N-subtiles of 64
  int rowBlk = blockIdx.y * 128;
  int colBlk = blockIdx.x * 128;

  // cooperative fill: thread -> row = tid/2, 32B half-row (tid&1), 2 x b128 per matrix
  int fRow   = tid >> 1;
  int fElem  = (tid & 1) * 16;      // element offset inside the 32-elem slab row
  const bf16_t* gArow = A  + (size_t)(rowBlk + fRow) * K + fElem;
  const bf16_t* gBrow = Bm + (size_t)(colBlk + fRow) * K + fElem;
  unsigned ldsArow[2], ldsBrow[2];
#pragma unroll
  for (int bufi = 0; bufi < 2; ++bufi) {
    ldsArow[bufi] = lds_addr32(&ldsA[bufi][fRow * PADLD + fElem]);
    ldsBrow[bufi] = lds_addr32(&ldsB[bufi][fRow * PADLD + fElem]);
  }

  auto fill = [&](int bufi, int kb) {
#pragma unroll
    for (int c = 0; c < 2; ++c) {
      async_copy_b128(ldsArow[bufi] + c * 16, gArow + kb + c * 8);
      async_copy_b128(ldsBrow[bufi] + c * 16, gBrow + kb + c * 8);
    }
  };

  v8f acc[2][4] = {};

  fill(0, 0);
  wait_async0();
  __syncthreads();

  for (int kb = 0; kb < K; kb += KTILE) {
    int cur = (kb / KTILE) & 1;
    if (kb + KTILE < K) fill(1 - cur, kb + KTILE);   // prefetch next slab

    const unsigned short* bufA = &ldsA[cur][0];
    const unsigned short* bufB = &ldsB[cur][0];
    v16bf a0 = load_frag_lds(bufA, wm * 32,      lane);
    v16bf a1 = load_frag_lds(bufA, wm * 32 + 16, lane);
    v16bf b0 = load_frag_lds(bufB, wn * 64,      lane);
    v16bf b1 = load_frag_lds(bufB, wn * 64 + 16, lane);
    v16bf b2 = load_frag_lds(bufB, wn * 64 + 32, lane);
    v16bf b3 = load_frag_lds(bufB, wn * 64 + 48, lane);
    acc[0][0] = wmma_bf16(a0, b0, acc[0][0]);
    acc[0][1] = wmma_bf16(a0, b1, acc[0][1]);
    acc[0][2] = wmma_bf16(a0, b2, acc[0][2]);
    acc[0][3] = wmma_bf16(a0, b3, acc[0][3]);
    acc[1][0] = wmma_bf16(a1, b0, acc[1][0]);
    acc[1][1] = wmma_bf16(a1, b1, acc[1][1]);
    acc[1][2] = wmma_bf16(a1, b2, acc[1][2]);
    acc[1][3] = wmma_bf16(a1, b3, acc[1][3]);

    wait_async0();       // own prefetches landed
    __syncthreads();     // everyone's prefetches landed; slab reads complete
  }

  int laneHi = lane >> 4;
  int nIn    = lane & 15;
#pragma unroll
  for (int i = 0; i < 2; ++i) {
#pragma unroll
    for (int j = 0; j < 4; ++j) {
      int col  = colBlk + wn * 64 + j * 16 + nIn;
      float bv = bias ? bias[col] : 0.f;
#pragma unroll
      for (int r = 0; r < 8; ++r) {
        int m   = rowBlk + wm * 32 + i * 16 + r + laneHi * 8;
        float v = acc[i][j][r] + bv;
        if (residual) v += residual[(size_t)m * N + col];
        out[(size_t)m * N + col] = v;
      }
    }
  }
}

// ---------------- kernel 4: RoPE(q,k) + split + V transpose ----------------
// q gets the 1/sqrt(D)=1/8 score scale folded in. vT layout: [bh][d][l].
__global__ void rope_split_kernel(const float* __restrict__ qkv,
                                  bf16_t* __restrict__ q,
                                  bf16_t* __restrict__ k,
                                  bf16_t* __restrict__ vT) {
  int idx = blockIdx.x * blockDim.x + threadIdx.x;   // B*H*L*32 threads exactly
  int j   = idx & 31;
  int l   = (idx >> 5) & (L_DIM - 1);
  int bh  = idx >> 16;                               // 0..31
  int h   = bh & (H_DIM - 1);
  int b   = bh >> 4;
  const float* src = qkv + (size_t)(b * L_DIM + l) * QKV_N + h * D_DIM;

  float invf = __powf(10000.f, -(float)j * (1.0f / 32.0f));
  float ang  = (float)l * invf;
  float sn, cs;
  __sincosf(ang, &sn, &cs);

  float q1 = src[j],          q2 = src[j + 32];
  float k1 = src[C_DIM + j],  k2 = src[C_DIM + j + 32];
  size_t ob = ((size_t)bh * L_DIM + l) * D_DIM;
  q[ob + j]      = f2bf((q1 * cs - q2 * sn) * 0.125f);
  q[ob + j + 32] = f2bf((q2 * cs + q1 * sn) * 0.125f);
  k[ob + j]      = f2bf(k1 * cs - k2 * sn);
  k[ob + j + 32] = f2bf(k2 * cs + k1 * sn);

  float v1 = src[2 * C_DIM + j], v2 = src[2 * C_DIM + j + 32];
  size_t vb = (size_t)bh * D_DIM * L_DIM;
  vT[vb + (size_t)j * L_DIM + l]        = f2bf(v1);
  vT[vb + (size_t)(j + 32) * L_DIM + l] = f2bf(v2);
}

// ---------------- kernel 5: flash attention + gelu + cat ----------------
// grid (B*H, L/128); 8 waves, each wave owns 16 q-rows, 32 keys per iter.
__global__ __launch_bounds__(256) void attn_kernel(const bf16_t* __restrict__ q,
                                                   const bf16_t* __restrict__ k,
                                                   const bf16_t* __restrict__ vT,
                                                   bf16_t* __restrict__ cat) {
  int bh   = blockIdx.x;
  int tid  = threadIdx.x;
  int lane = tid & 31;
  int w    = tid >> 5;
  int laneHi = lane >> 4;
  int nIn    = lane & 15;
  int h = bh & (H_DIM - 1);
  int b = bh >> 4;

  const bf16_t* qb = q  + (size_t)bh * L_DIM * D_DIM;
  const bf16_t* kb = k  + (size_t)bh * L_DIM * D_DIM;
  const bf16_t* vb = vT + (size_t)bh * D_DIM * L_DIM;
  int q0 = blockIdx.y * 128 + w * 16;

  v16bf qa0 = load_frag_g(qb, q0, 0,  D_DIM, lane);   // K dims 0..31
  v16bf qa1 = load_frag_g(qb, q0, 32, D_DIM, lane);   // K dims 32..63

  v8f o[4] = {};
  float rowM[8], rowL[8];
#pragma unroll
  for (int r = 0; r < 8; ++r) { rowM[r] = -3.0e38f; rowL[r] = 0.f; }

  __shared__ unsigned short plds[8][16 * 32];         // per-wave P tile
  unsigned short* pl = &plds[w][0];

  for (int kt = 0; kt < L_DIM; kt += 32) {
    v8f s0 = {}, s1 = {};
    v16bf kf0a = load_frag_g(kb, kt,       0,  D_DIM, lane);
    v16bf kf0b = load_frag_g(kb, kt,       32, D_DIM, lane);
    v16bf kf1a = load_frag_g(kb, kt + 16,  0,  D_DIM, lane);
    v16bf kf1b = load_frag_g(kb, kt + 16,  32, D_DIM, lane);
    s0 = wmma_bf16(qa0, kf0a, s0);
    s0 = wmma_bf16(qa1, kf0b, s0);
    s1 = wmma_bf16(qa0, kf1a, s1);
    s1 = wmma_bf16(qa1, kf1b, s1);

    // online softmax: row r lives at (vgpr r, lanes of this half)
#pragma unroll
    for (int r = 0; r < 8; ++r) {
      float mx = fmaxf(s0[r], s1[r]);
#pragma unroll
      for (int msk = 1; msk < 16; msk <<= 1) mx = fmaxf(mx, __shfl_xor(mx, msk, 32));
      float newM  = fmaxf(rowM[r], mx);
      float alpha = __expf(rowM[r] - newM);
      float p0 = __expf(s0[r] - newM);
      float p1 = __expf(s1[r] - newM);
      float ps = p0 + p1;
#pragma unroll
      for (int msk = 1; msk < 16; msk <<= 1) ps += __shfl_xor(ps, msk, 32);
      rowL[r] = rowL[r] * alpha + ps;
      rowM[r] = newM;
#pragma unroll
      for (int j = 0; j < 4; ++j) o[j][r] *= alpha;
      int mLoc = r + laneHi * 8;
      pl[mLoc * 32 + nIn]      = __builtin_bit_cast(unsigned short, f2bf(p0));
      pl[mLoc * 32 + 16 + nIn] = __builtin_bit_cast(unsigned short, f2bf(p1));
    }
    asm volatile("s_wait_dscnt 0" ::: "memory");

    // re-read P (16x32) in A-fragment layout from LDS (wave-private tile)
    int rr = lane & 15;
    int ko = (lane >> 4) << 3;
    const unsigned short* basep = pl + rr * 32 + ko;
    v8u16 lo = *(const v8u16*)(basep);
    v8u16 hi = *(const v8u16*)(basep + 16);
    v16bf pa;
#pragma unroll
    for (int e = 0; e < 8; ++e) {
      pa[e]     = __builtin_bit_cast(bf16_t, (unsigned short)lo[e]);
      pa[e + 8] = __builtin_bit_cast(bf16_t, (unsigned short)hi[e]);
    }
#pragma unroll
    for (int j = 0; j < 4; ++j) {
      v16bf vf = load_frag_g(vb, j * 16, kt, L_DIM, lane);   // B = V[32k x 16d]
      o[j] = wmma_bf16(pa, vf, o[j]);
    }
  }

  // epilogue: write o and gelu(o) into cat
#pragma unroll
  for (int j = 0; j < 4; ++j) {
    int d = j * 16 + nIn;
#pragma unroll
    for (int r = 0; r < 8; ++r) {
      int   ql = q0 + r + laneHi * 8;
      float ov = o[j][r] / rowL[r];
      float g  = 0.5f * ov *
                 (1.f + tanhf(0.7978845608f * (ov + 0.044715f * ov * ov * ov)));
      size_t ro = (size_t)(b * L_DIM + ql) * CAT_N;
      cat[ro + h * D_DIM + d]         = f2bf(ov);
      cat[ro + C_DIM + h * D_DIM + d] = f2bf(g);
    }
  }
}

// ---------------- launcher ----------------
extern "C" void kernel_launch(void* const* d_in, const int* in_sizes, int n_in,
                              void* d_out, int out_size, void* d_ws, size_t ws_size,
                              hipStream_t stream) {
  const float* x    = (const float*)d_in[0];
  const float* ln_w = (const float*)d_in[1];
  const float* ln_b = (const float*)d_in[2];
  const float* w1   = (const float*)d_in[3];
  const float* b1   = (const float*)d_in[4];
  const float* w2   = (const float*)d_in[5];
  const float* b2   = (const float*)d_in[6];
  float* out = (float*)d_out;
  (void)in_sizes; (void)n_in; (void)out_size; (void)ws_size;

  char* p = (char*)d_ws;
  auto take = [&](size_t bytes) { char* r = p; p += (bytes + 255) & ~(size_t)255; return r; };

  bf16_t* h_bf   = (bf16_t*)take((size_t)NROWS * C_DIM * 2);          // 8 MB
  bf16_t* w1_bf  = (bf16_t*)take((size_t)QKV_N * C_DIM * 2);          // 6 MB
  bf16_t* w2_bf  = (bf16_t*)take((size_t)C_DIM * CAT_N * 2);          // 4 MB
  float*  qkv    = (float*) take((size_t)NROWS * QKV_N * 4);          // 48 MB
  bf16_t* q_bf   = (bf16_t*)take((size_t)B_DIM * H_DIM * L_DIM * D_DIM * 2); // 8 MB
  bf16_t* k_bf   = (bf16_t*)take((size_t)B_DIM * H_DIM * L_DIM * D_DIM * 2); // 8 MB
  bf16_t* vT_bf  = (bf16_t*)take((size_t)B_DIM * H_DIM * L_DIM * D_DIM * 2); // 8 MB
  bf16_t* cat_bf = (bf16_t*)take((size_t)NROWS * CAT_N * 2);          // 16 MB

  // 1) LayerNorm + bf16 cast
  ln_kernel<<<NROWS, 256, 0, stream>>>(x, ln_w, ln_b, h_bf);
  // 2) weight casts (only live rows of w1)
  cast_bf16_kernel<<<(QKV_N * C_DIM + 255) / 256, 256, 0, stream>>>(w1, w1_bf, QKV_N * C_DIM);
  cast_bf16_kernel<<<(C_DIM * CAT_N + 255) / 256, 256, 0, stream>>>(w2, w2_bf, C_DIM * CAT_N);
  // 3) GEMM1: qkv = h * w1[:3C]^T + b1
  gemm_bf16_kernel<<<dim3(QKV_N / 128, NROWS / 128), 256, 0, stream>>>(
      h_bf, w1_bf, b1, nullptr, qkv, NROWS, QKV_N, C_DIM);
  // 4) RoPE + head split + V transpose
  rope_split_kernel<<<(B_DIM * H_DIM * L_DIM * 32) / 256, 256, 0, stream>>>(
      qkv, q_bf, k_bf, vT_bf);
  // 5) flash attention + gelu + concat
  attn_kernel<<<dim3(B_DIM * H_DIM, L_DIM / 128), 256, 0, stream>>>(
      q_bf, k_bf, vT_bf, cat_bf);
  // 6) GEMM2: out = cat * w2^T + b2 + x
  gemm_bf16_kernel<<<dim3(C_DIM / 128, NROWS / 128), 256, 0, stream>>>(
      cat_bf, w2_bf, b2, x, out, NROWS, C_DIM, CAT_N);
}
